// Transformer_aggregation_encoder_6914897346692
// MI455X (gfx1250) — matrix operand order repaired and checked by reading, not compiled, not run on hardware
//
#include <hip/hip_runtime.h>
#include <hip/hip_bf16.h>

// ---------------------------------------------------------------------------
// Problem constants (from reference): L=6, T=128, B=256, D=512, H=8, DFF=2048
// ---------------------------------------------------------------------------
#define LNUM 6
#define TT   128
#define BB   256
#define DD   512
#define HH   8
#define DFF  2048
#define SUBD 64
#define BT   (BB * TT)          // 32768 rows for all big GEMMs
#define TBD  (TT * BB * DD)     // 16,777,216 elements

typedef unsigned short u16;
typedef __attribute__((ext_vector_type(16))) __bf16 v16bf;
typedef __attribute__((ext_vector_type(8)))  float  v8f;

struct U8x32 { unsigned int u[8]; };   // 32 bytes == v16bf

// --------------------------- bf16 helpers ----------------------------------
__device__ __forceinline__ u16 f2bf(float x) {
    unsigned int u = __float_as_uint(x);
    return (u16)((u + 0x7FFFu + ((u >> 16) & 1u)) >> 16);   // round-nearest-even
}
__device__ __forceinline__ float bf2f(u16 h) {
    return __uint_as_float(((unsigned int)h) << 16);
}

// ---- CDNA5 async memory->LDS copy (ASYNCcnt-tracked, gfx1250) --------------
// Generic pointers to __shared__ carry the LDS byte offset in their low 32 bits.
__device__ __forceinline__ unsigned lds_off(const void* p) {
    return (unsigned)(unsigned long long)p;
}
__device__ __forceinline__ void async_copy_b128(unsigned lds_dst, const void* gsrc) {
    asm volatile("global_load_async_to_lds_b128 %0, %1, off"
                 :: "v"(lds_dst), "v"(gsrc) : "memory");
}
template <int N>
__device__ __forceinline__ void wait_async() {
    asm volatile("s_wait_asynccnt %0" :: "i"(N) : "memory");
}

// ---------------------------------------------------------------------------
// Fragment loaders matching CDNA5 wave32 WMMA VGPR layouts (ISA 7.12.2).
// A (16x32, bf16): lanes 0-15 -> M=lane, K groups {0..7,16..23};
//                  lanes 16-31 -> same M, K groups {8..15,24..31}.
// B (32x16, bf16): lanes 0-15 -> N=lane, K groups {0..7,16..23}; lanes 16-31 -> K groups {8..15,24..31}.
// Consecutive-K pairs share one dword -> 8 x 32-bit LDS reads per fragment.
// ---------------------------------------------------------------------------
__device__ __forceinline__ v16bf load_frag_a(const u16* lds, int rowBase,
                                             int strideH, int lane) {
    const int lrow = lane & 15;
    const int kb   = (lane >> 4) * 8;
    const u16* p = lds + (rowBase + lrow) * strideH + kb;
    U8x32 f;
#pragma unroll
    for (int j = 0; j < 8; ++j) {
        const int off = (j < 4) ? (2 * j) : (8 + 2 * j);    // 0,2,4,6,16,18,20,22
        f.u[j] = *(const unsigned int*)(p + off);
    }
    return __builtin_bit_cast(v16bf, f);
}

__device__ __forceinline__ v16bf load_frag_b(const u16* lds, int colBase,
                                             int strideH, int lane) {
    const int lcol = lane & 15;
    const int kb   = (lane >> 4) * 8;
    const u16* p = lds + (colBase + lcol) * strideH + kb;
    U8x32 f;
#pragma unroll
    for (int j = 0; j < 8; ++j) f.u[j] = *(const unsigned int*)(p + ((j < 4) ? (2 * j) : (8 + 2 * j)));
    return __builtin_bit_cast(v16bf, f);
}

// ---------------------------------------------------------------------------
// Generic tiled WMMA GEMM:  C[m,n] = act( A[m,:] . W[n,:] + bias[n] )
// A: [M,K] bf16 row-major, W: [N,K] bf16 row-major (einsum '...d,ed->...e').
// Block tile 128(M) x 256(N), 8 waves as 2(M) x 4(N); wave tile 64x64 =
// 4x4 fragments -> 16 v_wmma_f32_16x16x32_bf16 per 8 fragment loads per K-step.
// Double-buffered LDS pipeline: tile k+1 streams in via
// GLOBAL_LOAD_ASYNC_TO_LDS_B128 while tile k is consumed by WMMA; async loads
// complete in order, so s_wait_asynccnt 6 retires exactly the current tile.
// ---------------------------------------------------------------------------
#define MBLK 128
#define NBLK 256
#define KBLK 32
#define LDH  40   // halves: 80B rows -> 16B-aligned for async b128, 20-dword stride

template <bool RELU, bool OUT_BF16>
__global__ __launch_bounds__(256) void gemm_bf16_kernel(
    const u16* __restrict__ A, const u16* __restrict__ W,
    const float* __restrict__ bias, void* __restrict__ Cout,
    int M, int N, int K) {
    __shared__ alignas(16) u16 lA[2][MBLK * LDH];
    __shared__ alignas(16) u16 lW[2][NBLK * LDH];

    const int tid   = threadIdx.x;
    const int lane  = tid & 31;
    const int wave  = tid >> 5;
    const int waveM = wave >> 2;   // 0..1 -> 64-row slice
    const int waveN = wave & 3;    // 0..3 -> 64-col slice
    const int bm = blockIdx.y * MBLK;
    const int bn = blockIdx.x * NBLK;

    v8f acc[4][4] = {};

    const int ldRow = tid >> 1;          // 0..127
    const int ldSeg = (tid & 1) * 16;    // halves (0 or 32B)

    const u16* gA  = A + (size_t)(bm + ldRow) * K + ldSeg;
    const u16* gW0 = W + (size_t)(bn + ldRow) * K + ldSeg;
    const u16* gW1 = W + (size_t)(bn + 128 + ldRow) * K + ldSeg;

    unsigned aDst[2], wDst0[2], wDst1[2];
#pragma unroll
    for (int p = 0; p < 2; ++p) {
        aDst[p]  = lds_off(&lA[p][ldRow * LDH + ldSeg]);
        wDst0[p] = lds_off(&lW[p][ldRow * LDH + ldSeg]);
        wDst1[p] = lds_off(&lW[p][(128 + ldRow) * LDH + ldSeg]);
    }

    // 96B per thread per tile: A 128x32, W 256x32 halves = 6 async b128 each.
    auto issue_tile = [&](int k0, int p) {
        async_copy_b128(aDst[p],       gA + k0);
        async_copy_b128(aDst[p] + 16,  gA + k0 + 8);
        async_copy_b128(wDst0[p],      gW0 + k0);
        async_copy_b128(wDst0[p] + 16, gW0 + k0 + 8);
        async_copy_b128(wDst1[p],      gW1 + k0);
        async_copy_b128(wDst1[p] + 16, gW1 + k0 + 8);
    };

    issue_tile(0, 0);                      // prologue: tile 0 -> buffer 0
    const int nk = K / KBLK;
    for (int kt = 0; kt < nk; ++kt) {
        const int cur = kt & 1;
        if (kt + 1 < nk) {
            issue_tile((kt + 1) * KBLK, cur ^ 1);   // overlap next tile's DMA
            wait_async<6>();   // in-order completion => tile kt has landed
        } else {
            wait_async<0>();
        }
        if (kt + 2 < nk) {     // keep L2 ahead of the DMA (global_prefetch_b8)
            __builtin_prefetch(gA  + (kt + 2) * KBLK, 0, 1);
            __builtin_prefetch(gW0 + (kt + 2) * KBLK, 0, 1);
            __builtin_prefetch(gW1 + (kt + 2) * KBLK, 0, 1);
        }
        __syncthreads();       // all waves' slices of tile kt visible

        v16bf bfrag[4];
#pragma unroll
        for (int j = 0; j < 4; ++j)
            bfrag[j] = load_frag_b(lW[cur], waveN * 64 + j * 16, LDH, lane);
#pragma unroll
        for (int i = 0; i < 4; ++i) {
            v16bf afrag = load_frag_a(lA[cur], waveM * 64 + i * 16, LDH, lane);
#pragma unroll
            for (int j = 0; j < 4; ++j)
                acc[i][j] = __builtin_amdgcn_wmma_f32_16x16x32_bf16(
                    false, afrag, false, bfrag[j], (short)0, acc[i][j], false, false);
        }
        __syncthreads();       // done reading buffer `cur` before it is refilled
    }

    // Epilogue: C/D layout -> VGPR r: M = r + 8*(lane>=16), N = lane&15
    const int lrow = lane & 15;
    const int lk8  = (lane >> 4) * 8;
#pragma unroll
    for (int i = 0; i < 4; ++i) {
#pragma unroll
        for (int j = 0; j < 4; ++j) {
            const int gn = bn + waveN * 64 + j * 16 + lrow;
            const float bv = bias[gn];
#pragma unroll
            for (int r = 0; r < 8; ++r) {
                const int gm = bm + waveM * 64 + i * 16 + lk8 + r;
                float v = acc[i][j][r] + bv;
                if (RELU) v = v > 0.f ? v : 0.f;
                if (OUT_BF16)
                    ((u16*)Cout)[(size_t)gm * N + gn] = f2bf(v);
                else
                    ((float*)Cout)[(size_t)gm * N + gn] = v;
            }
        }
    }
}

// ---------------------------------------------------------------------------
// Per-head message GEMM, fully fused:
//   prod[r,s] = s[r, h*64+s] * t[r, h*64+s]           (built into LDS)
//   msg[r,u]  = sum_s prod[r,s] * mw[h,u,s] + mb[h,u] (WMMA, K=64)
//   wm[b,t,h*64+u] = relu( wts[h,b,t] * msg[r,u] )    (r = t*B + b)
// One block: one head x 128 rows; 8 waves each own 16 rows x 64 cols.
// ---------------------------------------------------------------------------
#define MLDH 68

__global__ __launch_bounds__(256) void msg_kernel(
    const u16* __restrict__ Sb, const u16* __restrict__ Tb,
    const u16* __restrict__ MW, const float* __restrict__ mb,
    const float* __restrict__ wts, u16* __restrict__ wm) {
    __shared__ u16 lP[128 * MLDH];
    __shared__ u16 lW[64 * MLDH];

    const int h    = blockIdx.y;
    const int rb   = blockIdx.x * 128;
    const int tid  = threadIdx.x;
    const int lane = tid & 31;
    const int wave = tid >> 5;

    {   // build product tile: 128 rows x 64 (one head's sub-slice)
        const int row = tid >> 1;
        const int seg = (tid & 1) * 32;
        const size_t g = (size_t)(rb + row) * DD + h * SUBD + seg;
#pragma unroll
        for (int j = 0; j < 32; ++j)
            lP[row * MLDH + seg + j] = f2bf(bf2f(Sb[g + j]) * bf2f(Tb[g + j]));
    }
    {   // mw tile: lW[u][s], 64x64
        const int u   = tid >> 2;
        const int seg = (tid & 3) * 16;
        const u16* gp = MW + (size_t)h * SUBD * SUBD + u * SUBD + seg;
#pragma unroll
        for (int j = 0; j < 16; ++j) lW[u * MLDH + seg + j] = gp[j];
    }
    __syncthreads();

    v8f acc[4] = {};
#pragma unroll
    for (int kk = 0; kk < 2; ++kk) {          // K = 64 = 2 x 32
        v16bf af = load_frag_a(lP + kk * 32, wave * 16, MLDH, lane);
#pragma unroll
        for (int j = 0; j < 4; ++j) {
            v16bf bf = load_frag_b(lW + kk * 32, j * 16, MLDH, lane);
            acc[j] = __builtin_amdgcn_wmma_f32_16x16x32_bf16(
                false, af, false, bf, (short)0, acc[j], false, false);
        }
    }

    const int lrow = lane & 15;
    const int lk8  = (lane >> 4) * 8;
#pragma unroll
    for (int j = 0; j < 4; ++j) {
#pragma unroll
        for (int r = 0; r < 8; ++r) {
            const int gm = rb + wave * 16 + lk8 + r;   // r = t*B + b
            const int u  = j * 16 + lrow;
            const int tt = gm >> 8;        // / B
            const int b  = gm & 255;       // % B
            float v = acc[j][r] + mb[h * SUBD + u];
            v *= wts[((size_t)h * BB + b) * TT + tt];
            v = v > 0.f ? v : 0.f;
            wm[((size_t)b * TT + tt) * DD + h * SUBD + u] = f2bf(v);
        }
    }
}

// ---------------------------------------------------------------------------
// scores[h,b,t] = sum_s s[t,b,h*64+s]*aw[h,s] + sum_s t[..]*aw[h,64+s] + ab[h]
// ---------------------------------------------------------------------------
__global__ __launch_bounds__(256) void scores_kernel(
    const u16* __restrict__ Sb, const u16* __restrict__ Tb,
    const float* __restrict__ aw, const float* __restrict__ ab,
    float* __restrict__ scores) {
    const int idx = blockIdx.x * 256 + threadIdx.x;   // h*B*T
    const int h   = idx >> 15;
    const int rem = idx & 32767;
    const int b   = rem >> 7;
    const int t   = rem & 127;
    const size_t base = ((size_t)t * BB + b) * DD + h * SUBD;
    const float* awh = aw + h * (2 * SUBD);
    float acc = ab[h];
#pragma unroll 8
    for (int s = 0; s < SUBD; ++s) {
        acc += bf2f(Sb[base + s]) * awh[s];
        acc += bf2f(Tb[base + s]) * awh[SUBD + s];
    }
    scores[idx] = acc;   // layout [h,b,t]
}

// softmax over the batch axis (B=256 == blockDim)
__global__ __launch_bounds__(256) void softmax_b_kernel(
    const float* __restrict__ scores, float* __restrict__ wts) {
    __shared__ float red[256];
    const int h = blockIdx.x >> 7;
    const int t = blockIdx.x & 127;
    const int b = threadIdx.x;
    const size_t idx = ((size_t)h * BB + b) * TT + t;
    const float v = scores[idx];
    red[b] = v; __syncthreads();
    for (int s = 128; s > 0; s >>= 1) {
        if (b < s) red[b] = fmaxf(red[b], red[b + s]);
        __syncthreads();
    }
    const float mx = red[0]; __syncthreads();
    const float e = __expf(v - mx);
    red[b] = e; __syncthreads();
    for (int s = 128; s > 0; s >>= 1) {
        if (b < s) red[b] += red[b + s];
        __syncthreads();
    }
    wts[idx] = e / red[0];
}

// ---------------------------------------------------------------------------
// Fused residual-add + LayerNorm over D=512 (one block per row, 2 elems/thread)
// Writes normalized f32 in place; optionally a bf16 copy for the next GEMM.
// ---------------------------------------------------------------------------
template <bool WRITE_BF16>
__global__ __launch_bounds__(256) void layernorm_kernel(
    float* __restrict__ X, const float* __restrict__ Y,
    const float* __restrict__ g, const float* __restrict__ bta,
    u16* __restrict__ Xbf) {
    __shared__ float red[256];
    const size_t row = blockIdx.x;
    const int tid = threadIdx.x;
    const size_t o0 = row * DD + tid;
    const size_t o1 = o0 + 256;
    float v0 = X[o0] + Y[o0];
    float v1 = X[o1] + Y[o1];
    red[tid] = v0 + v1; __syncthreads();
    for (int s = 128; s > 0; s >>= 1) {
        if (tid < s) red[tid] += red[tid + s];
        __syncthreads();
    }
    const float mean = red[0] * (1.f / DD); __syncthreads();
    const float d0 = v0 - mean, d1 = v1 - mean;
    red[tid] = d0 * d0 + d1 * d1; __syncthreads();
    for (int s = 128; s > 0; s >>= 1) {
        if (tid < s) red[tid] += red[tid + s];
        __syncthreads();
    }
    const float inv = rsqrtf(red[0] * (1.f / DD) + 1e-5f);
    const float r0 = d0 * inv * g[tid] + bta[tid];
    const float r1 = d1 * inv * g[256 + tid] + bta[256 + tid];
    X[o0] = r0; X[o1] = r1;
    if (WRITE_BF16) { Xbf[o0] = f2bf(r0); Xbf[o1] = f2bf(r1); }
}

// --------------------------- layout shuffles -------------------------------
__global__ __launch_bounds__(256) void f32_to_bf16_kernel(
    const float* __restrict__ in, u16* __restrict__ out) {
    const size_t i = (size_t)blockIdx.x * 256 + threadIdx.x;
    out[i] = f2bf(in[i]);
}
// [B,T,D] f32 -> [T,B,D] bf16
__global__ __launch_bounds__(256) void bt2tb_bf16_kernel(
    const float* __restrict__ X, u16* __restrict__ out) {
    const size_t idx = (size_t)blockIdx.x * 256 + threadIdx.x;
    const int d = (int)(idx & 511);
    const size_t r = idx >> 9;             // b*T + t
    const int t = (int)(r & 127);
    const size_t b = r >> 7;
    out[(((size_t)t * BB + b) << 9) + d] = f2bf(X[idx]);
}
// [T,B,D] f32 -> [B,T,D] f32
__global__ __launch_bounds__(256) void tb2bt_f32_kernel(
    const float* __restrict__ src, float* __restrict__ out) {
    const size_t idx = (size_t)blockIdx.x * 256 + threadIdx.x;
    const int d = (int)(idx & 511);
    const size_t r = idx >> 9;             // t*B + b
    const int b = (int)(r & 255);
    const size_t t = r >> 8;
    out[(((size_t)b * TT + t) << 9) + d] = src[idx];
}
// [B,T,D] f32 -> [T,B,D] f32 (final output)
__global__ __launch_bounds__(256) void bt2tb_f32_kernel(
    const float* __restrict__ X, float* __restrict__ out) {
    const size_t idx = (size_t)blockIdx.x * 256 + threadIdx.x;
    const int d = (int)(idx & 511);
    const size_t r = idx >> 9;             // b*T + t
    const int t = (int)(r & 127);
    const size_t b = r >> 7;
    out[(((size_t)t * BB + b) << 9) + d] = X[idx];
}

// ---------------------------------------------------------------------------
extern "C" void kernel_launch(void* const* d_in, const int* in_sizes, int n_in,
                              void* d_out, int out_size, void* d_ws, size_t ws_size,
                              hipStream_t stream) {
    const float* src       = (const float*)d_in[0];
    const float* srcc      = (const float*)d_in[1];
    const float* spatio_w  = (const float*)d_in[2];
    const float* spatio_b  = (const float*)d_in[3];
    const float* temporal_w= (const float*)d_in[4];
    const float* temporal_b= (const float*)d_in[5];
    const float* attn_w    = (const float*)d_in[6];
    const float* attn_b    = (const float*)d_in[7];
    const float* msg_w     = (const float*)d_in[8];
    const float* msg_b     = (const float*)d_in[9];
    const float* agg_w     = (const float*)d_in[10];
    const float* agg_b     = (const float*)d_in[11];
    const float* lin1_w    = (const float*)d_in[12];
    const float* lin1_b    = (const float*)d_in[13];
    const float* lin2_w    = (const float*)d_in[14];
    const float* lin2_b    = (const float*)d_in[15];
    const float* ln1_g     = (const float*)d_in[16];
    const float* ln1_b     = (const float*)d_in[17];
    const float* ln2_g     = (const float*)d_in[18];
    const float* ln2_b     = (const float*)d_in[19];
    float* out = (float*)d_out;

    // ---- workspace carve (256B aligned) ----
    char* ws = (char*)d_ws;
    auto carve = [&](size_t bytes) {
        char* p = ws;
        ws += (bytes + 255) & ~(size_t)255;
        return p;
    };
    u16*   srcc_bf = (u16*)  carve((size_t)TBD * 2);
    u16*   xbf     = (u16*)  carve((size_t)TBD * 2);  // [T,B,D] then [B,T,D]
    u16*   s_buf   = (u16*)  carve((size_t)TBD * 2);
    u16*   tp_buf  = (u16*)  carve((size_t)TBD * 2);
    u16*   wm      = (u16*)  carve((size_t)TBD * 2);
    u16*   h1      = (u16*)  carve((size_t)BT * DFF * 2);
    float* F0      = (float*)carve((size_t)TBD * 4);  // running x [B,T,D]
    float* F1      = (float*)carve((size_t)TBD * 4);  // agg / y
    float* scores  = (float*)carve((size_t)HH * BB * TT * 4);
    float* wts     = (float*)carve((size_t)HH * BB * TT * 4);
    u16*   wbf     = (u16*)  carve((size_t)DFF * DD * 2);  // current weight (bf16)
    u16*   mwbf    = (u16*)  carve((size_t)HH * SUBD * SUBD * 2);

    const int cvtTBD = TBD / 256;          // 65536 blocks
    const int cvtDxD = (DD * DD) / 256;
    const int cvtDFF = (DFF * DD) / 256;
    const int cvtMW  = (HH * SUBD * SUBD) / 256;

    // ---- one-time setup ----
    f32_to_bf16_kernel<<<cvtTBD, 256, 0, stream>>>(srcc, srcc_bf);
    tb2bt_f32_kernel<<<cvtTBD, 256, 0, stream>>>(src, F0);

    for (int l = 0; l < LNUM; ++l) {
        const float* sw  = spatio_w  + (size_t)l * DD * DD;
        const float* sb  = spatio_b  + (size_t)l * DD;
        const float* tw  = temporal_w+ (size_t)l * DD * DD;
        const float* tb  = temporal_b+ (size_t)l * DD;
        const float* aw  = attn_w    + (size_t)l * HH * 2 * SUBD;
        const float* ab  = attn_b    + (size_t)l * HH;
        const float* mw  = msg_w     + (size_t)l * HH * SUBD * SUBD;
        const float* mb  = msg_b     + (size_t)l * HH * SUBD;
        const float* gw  = agg_w     + (size_t)l * DD * DD;
        const float* gb  = agg_b     + (size_t)l * DD;
        const float* w1  = lin1_w    + (size_t)l * DFF * DD;
        const float* b1  = lin1_b    + (size_t)l * DFF;
        const float* w2  = lin2_w    + (size_t)l * DD * DFF;
        const float* b2  = lin2_b    + (size_t)l * DD;
        const float* n1g = ln1_g + (size_t)l * DD;
        const float* n1b = ln1_b + (size_t)l * DD;
        const float* n2g = ln2_g + (size_t)l * DD;
        const float* n2b = ln2_b + (size_t)l * DD;

        // s = relu(srcc @ sw^T + sb)   [T*B, 512]
        f32_to_bf16_kernel<<<cvtDxD, 256, 0, stream>>>(sw, wbf);
        gemm_bf16_kernel<true, true><<<dim3(DD / NBLK, BT / MBLK), 256, 0, stream>>>(
            srcc_bf, wbf, sb, s_buf, BT, DD, DD);

        // t = relu(x @ tw^T + tb); x in [T,B] order
        bt2tb_bf16_kernel<<<cvtTBD, 256, 0, stream>>>(F0, xbf);
        f32_to_bf16_kernel<<<cvtDxD, 256, 0, stream>>>(tw, wbf);
        gemm_bf16_kernel<true, true><<<dim3(DD / NBLK, BT / MBLK), 256, 0, stream>>>(
            xbf, wbf, tb, tp_buf, BT, DD, DD);

        // per-head scores + softmax over batch axis
        scores_kernel<<<(HH * BB * TT) / 256, 256, 0, stream>>>(s_buf, tp_buf, aw, ab, scores);
        softmax_b_kernel<<<HH * TT, 256, 0, stream>>>(scores, wts);

        // fused msg GEMM (prod build + WMMA + bias + weighting + relu)
        f32_to_bf16_kernel<<<cvtMW, 256, 0, stream>>>(mw, mwbf);
        msg_kernel<<<dim3(BT / 128, HH), 256, 0, stream>>>(s_buf, tp_buf, mwbf, mb, wts, wm);

        // agg = wm @ gw^T + gb  (f32 out)
        f32_to_bf16_kernel<<<cvtDxD, 256, 0, stream>>>(gw, wbf);
        gemm_bf16_kernel<false, false><<<dim3(DD / NBLK, BT / MBLK), 256, 0, stream>>>(
            wm, wbf, gb, F1, BT, DD, DD);

        // x = LN(x + agg); emit bf16 copy (xbf reused, now [B,T,D])
        layernorm_kernel<true><<<BT, 256, 0, stream>>>(F0, F1, n1g, n1b, xbf);

        // h1 = relu(x @ w1^T + b1)  [32768 x 2048]
        f32_to_bf16_kernel<<<cvtDFF, 256, 0, stream>>>(w1, wbf);
        gemm_bf16_kernel<true, true><<<dim3(DFF / NBLK, BT / MBLK), 256, 0, stream>>>(
            xbf, wbf, b1, h1, BT, DFF, DD);

        // y = h1 @ w2^T + b2
        f32_to_bf16_kernel<<<cvtDFF, 256, 0, stream>>>(w2, wbf);
        gemm_bf16_kernel<false, false><<<dim3(DD / NBLK, BT / MBLK), 256, 0, stream>>>(
            h1, wbf, b2, F1, BT, DD, DFF);

        // x = LN(x + y)
        layernorm_kernel<false><<<BT, 256, 0, stream>>>(F0, F1, n2g, n2b, nullptr);
    }

    // back to [T,B,D]
    bt2tb_f32_kernel<<<cvtTBD, 256, 0, stream>>>(F0, out);
}
